// SwarmLMBlock_4758823764524
// MI455X (gfx1250) — matrix operand order repaired and checked by reading, not compiled, not run on hardware
//
#include <hip/hip_runtime.h>
#include <math.h>

#define BQ 2
#define NQ 2048
#define DQ 1024
#define HQ 16
#define HDQ 64
#define CQ 32
#define WINQ 256

typedef _Float16 h16;
typedef __attribute__((ext_vector_type(16))) _Float16 v16h;
typedef __attribute__((ext_vector_type(8)))  float    v8f;

// A-fragment K index for 16x32 f16 A-matrix (ISA 7.12.2)
__device__ __forceinline__ int amap(int e, int half) {
  return (e & 7) + 8 * half + 16 * (e >> 3);
}

// Load one 32-deep K slab (16 floats of A + 16 of W per thread) into registers.
__device__ __forceinline__ void load_tile(
    const float* __restrict__ A, const float* __restrict__ A2,
    const float* __restrict__ Wt, int K1, int K2, int Ktot,
    int ar, int wrow, int lk, int k0, float4 ra[4], float4 rw[4])
{
  const float* srcA; int strideA, kbase;
  if (!A2 || k0 < K1) { srcA = A;  strideA = K1; kbase = k0; }
  else                { srcA = A2; strideA = K2; kbase = k0 - K1; }
  const float4* pa = (const float4*)(srcA + (size_t)ar * strideA + kbase + lk);
  const float4* pw = (const float4*)(Wt + (size_t)wrow * Ktot + k0 + lk);
  #pragma unroll
  for (int j = 0; j < 4; ++j) { ra[j] = pa[j]; rw[j] = pw[j]; }
}

// ---------------------------------------------------------------------------
// Generic WMMA GEMM:  Out[m,o] = act( sum_k A(m,k) * Wt[o,k] + bias[o] ) (+R)
// A(m,k): k < K1 -> A[m*K1+k]  else A2[m*(Ktot-K1)+(k-K1)]   (split for concat)
// Out is always [M, DQ].  act: 0=none, 1=gelu(exact), 2=sigmoid.
// Block: 256 threads (8 waves), tile 128x128; each wave owns 32x64
// (2 A-frags x 4 B-frags -> 8 WMMAs per K=32 step). K staged as f16 in LDS.
// Software-pipelined: next global tile is fetched into registers while the
// current tile's WMMAs execute between the two barriers.
// ---------------------------------------------------------------------------
__global__ __launch_bounds__(256) void gemm_kernel(
    const float* __restrict__ A, const float* __restrict__ A2,
    const float* __restrict__ Wt, const float* __restrict__ bias,
    const float* __restrict__ R, float* __restrict__ Out,
    int M, int K1, int Ktot, int act)
{
  __shared__ h16 As[128][40];
  __shared__ h16 Ws[128][40];
  const int tid  = threadIdx.x;
  const int lane = tid & 31, w = tid >> 5;
  const int r = lane & 15, half = lane >> 4;
  const int m0 = blockIdx.y * 128, o0 = blockIdx.x * 128;
  const int K2 = Ktot - K1;
  const int wr = (w & 3) * 32;   // wave row base within tile
  const int wc = (w >> 2) * 64;  // wave col base within tile
  const int lr = tid >> 1;       // staging row 0..127
  const int lk = (tid & 1) * 16; // staging k: 0 or 16

  v8f acc[2][4];
  #pragma unroll
  for (int mi = 0; mi < 2; ++mi)
    #pragma unroll
    for (int ni = 0; ni < 4; ++ni) acc[mi][ni] = (v8f){};

  int ar = m0 + lr; if (ar >= M) ar = M - 1;   // clamp for M < tile (small GEMMs)
  const int wrow = o0 + lr;                     // output-col row of Wt (always < DQ)

  float4 ra[4], rw[4];
  load_tile(A, A2, Wt, K1, K2, Ktot, ar, wrow, lk, 0, ra, rw);

  for (int k0 = 0; k0 < Ktot; k0 += 32) {
    // stage current slab (f32 -> f16) into LDS
    #pragma unroll
    for (int j = 0; j < 4; ++j) {
      As[lr][lk + 4 * j + 0] = (h16)ra[j].x; As[lr][lk + 4 * j + 1] = (h16)ra[j].y;
      As[lr][lk + 4 * j + 2] = (h16)ra[j].z; As[lr][lk + 4 * j + 3] = (h16)ra[j].w;
      Ws[lr][lk + 4 * j + 0] = (h16)rw[j].x; Ws[lr][lk + 4 * j + 1] = (h16)rw[j].y;
      Ws[lr][lk + 4 * j + 2] = (h16)rw[j].z; Ws[lr][lk + 4 * j + 3] = (h16)rw[j].w;
    }
    __syncthreads();

    // prefetch next slab into registers (overlaps with WMMAs below)
    const bool more = (k0 + 32) < Ktot;
    if (more) load_tile(A, A2, Wt, K1, K2, Ktot, ar, wrow, lk, k0 + 32, ra, rw);

    v16h a[2], bfr[4];
    #pragma unroll
    for (int mi = 0; mi < 2; ++mi)
      #pragma unroll
      for (int e = 0; e < 16; ++e) a[mi][e] = As[wr + mi * 16 + r][amap(e, half)];
    #pragma unroll
    for (int ni = 0; ni < 4; ++ni)
      #pragma unroll
      for (int e = 0; e < 16; ++e) bfr[ni][e] = Ws[wc + ni * 16 + r][amap(e, half)];
    #pragma unroll
    for (int mi = 0; mi < 2; ++mi)
      #pragma unroll
      for (int ni = 0; ni < 4; ++ni)
        acc[mi][ni] = __builtin_amdgcn_wmma_f32_16x16x32_f16(
            false, a[mi], false, bfr[ni], (short)0, acc[mi][ni], false, false);
    __syncthreads();
  }

  #pragma unroll
  for (int mi = 0; mi < 2; ++mi) {
    #pragma unroll
    for (int v = 0; v < 8; ++v) {
      int row = wr + mi * 16 + v + 8 * half;
      size_t m = (size_t)m0 + row;
      if ((int)m < M) {
        #pragma unroll
        for (int ni = 0; ni < 4; ++ni) {
          int o = o0 + wc + ni * 16 + r;
          float val = acc[mi][ni][v] + bias[o];
          if (act == 1)      val = 0.5f * val * (1.0f + erff(val * 0.70710678118f));
          else if (act == 2) val = 1.0f / (1.0f + expf(-val));
          if (R) val += R[m * DQ + o];
          Out[m * DQ + o] = val;
        }
      }
    }
  }
}

// ---------------------------------------------------------------------------
// Flash-style attention, 1 wave per (batch, head, 16-query block).
// HD = 64. mask_mode: 1 = causal + left window 256, 0 = full (global reps).
// Buffers are [B*Nrows, DQ] with head h at columns h*64..h*64+63.
// ---------------------------------------------------------------------------
__global__ __launch_bounds__(32) void attn_kernel(
    const float* __restrict__ Q, const float* __restrict__ K,
    const float* __restrict__ V, float* __restrict__ O,
    int Nrows, int mask_mode)
{
  const int qb = blockIdx.x, h = blockIdx.y, b = blockIdx.z;
  const int lane = threadIdx.x;
  const int r = lane & 15, half = lane >> 4;
  const int q0 = qb * 16;
  const size_t rowbase = (size_t)b * Nrows;
  const int hd0 = h * HDQ;

  __shared__ h16 Plds[16][18];

  // Q fragments (two K=32 chunks over HD=64), pre-scaled by 1/sqrt(HD)
  v16h qa[2];
  #pragma unroll
  for (int c = 0; c < 2; ++c)
    #pragma unroll
    for (int e = 0; e < 16; ++e) {
      int dd = c * 32 + amap(e, half);
      qa[c][e] = (h16)(Q[(rowbase + q0 + r) * DQ + hd0 + dd] * 0.125f);
    }

  v8f acc[4];
  #pragma unroll
  for (int t = 0; t < 4; ++t) acc[t] = (v8f){};
  float m_i[8], l_i[8];
  #pragma unroll
  for (int v = 0; v < 8; ++v) { m_i[v] = -1e30f; l_i[v] = 0.0f; }

  const int kb_lo = mask_mode ? ((q0 - WINQ > 0) ? (q0 - WINQ) >> 4 : 0) : 0;
  const int kb_hi = mask_mode ? qb : (Nrows >> 4) - 1;

  for (int kb = kb_lo; kb <= kb_hi; ++kb) {
    const int k0 = kb * 16;
    // S = Q K^T  (D-layout: row = v+8*half over queries, col = r over keys)
    v8f s = (v8f){};
    #pragma unroll
    for (int c = 0; c < 2; ++c) {
      v16h bfr;
      #pragma unroll
      for (int e = 0; e < 16; ++e) {
        int dd = c * 32 + amap(e, half);
        bfr[e] = (h16)K[(rowbase + k0 + r) * DQ + hd0 + dd];
      }
      s = __builtin_amdgcn_wmma_f32_16x16x32_f16(false, qa[c], false, bfr, (short)0, s, false, false);
    }

    const int j = k0 + r; // key column owned by this lane
    #pragma unroll
    for (int v = 0; v < 8; ++v) {
      const int i = q0 + v + 8 * half;
      float val = s[v];
      if (mask_mode && (j > i || j < i - WINQ)) val = -1e9f;
      // row max across the 16 lanes holding this row
      float mx = val;
      for (int off = 1; off < 16; off <<= 1) mx = fmaxf(mx, __shfl_xor(mx, off, 32));
      const float mnew = fmaxf(m_i[v], mx);
      const float scale = expf(m_i[v] - mnew);
      const float p = expf(val - mnew);
      float rs = p;
      for (int off = 1; off < 16; off <<= 1) rs += __shfl_xor(rs, off, 32);
      l_i[v] = l_i[v] * scale + rs;
      m_i[v] = mnew;
      #pragma unroll
      for (int t = 0; t < 4; ++t) acc[t][v] *= scale;
      Plds[v + 8 * half][r] = (h16)p;
    }
    __syncthreads();

    // P (16x16, zero-padded to K=32) in A layout
    v16h pa;
    #pragma unroll
    for (int e = 0; e < 16; ++e) {
      int kk = amap(e, half);
      pa[e] = (kk < 16) ? Plds[r][kk] : (h16)0.0f;
    }
    #pragma unroll
    for (int t = 0; t < 4; ++t) {
      v16h vb;
      #pragma unroll
      for (int e = 0; e < 16; ++e) {
        int kk = amap(e, half); // key index within padded K=32
        vb[e] = (kk < 16) ? (h16)V[(rowbase + k0 + kk) * DQ + hd0 + t * 16 + r]
                          : (h16)0.0f;
      }
      acc[t] = __builtin_amdgcn_wmma_f32_16x16x32_f16(false, pa, false, vb, (short)0, acc[t], false, false);
    }
    __syncthreads();
  }

  #pragma unroll
  for (int v = 0; v < 8; ++v) {
    const float inv = 1.0f / l_i[v];
    const int i = q0 + v + 8 * half;
    #pragma unroll
    for (int t = 0; t < 4; ++t)
      O[(rowbase + i) * DQ + hd0 + t * 16 + r] = acc[t][v] * inv;
  }
}

// ---------------------------------------------------------------------------
// Small elementwise / pooling kernels
// ---------------------------------------------------------------------------
__global__ void mean_kernel(const float* __restrict__ xl, float* __restrict__ reps) {
  int idx = blockIdx.x * blockDim.x + threadIdx.x; // over B*C*D
  int d = idx % DQ; int bc = idx / DQ; int c = bc % CQ; int b = bc / CQ;
  const float* p = xl + ((size_t)b * NQ + (size_t)c * 64) * DQ + d;
  float s = 0.0f;
  #pragma unroll 4
  for (int t = 0; t < 64; ++t) s += p[(size_t)t * DQ];
  reps[idx] = s * (1.0f / 64.0f);
}

__global__ void clustermix_kernel(const float* __restrict__ reps, float* __restrict__ mlin) {
  int idx = blockIdx.x * blockDim.x + threadIdx.x; // over B*C*D
  int d = idx % DQ; int bc = idx / DQ; int b = bc / CQ;
  float s = 0.0f;
  for (int c2 = 0; c2 < CQ; ++c2) s += reps[((size_t)b * CQ + c2) * DQ + d];
  mlin[idx] = s - reps[idx];
}

__global__ void bcast_kernel(const float* __restrict__ rnew, float* __restrict__ xg) {
  int idx = blockIdx.x * blockDim.x + threadIdx.x; // over B*N*D
  int d = idx % DQ; int bn = idx / DQ; int n = bn % NQ; int b = bn / NQ;
  int c = n / 64;
  xg[idx] = rnew[((size_t)b * CQ + c) * DQ + d];
}

__global__ void final_kernel(const float* __restrict__ xl, const float* __restrict__ xg,
                             const float* __restrict__ gate, float* __restrict__ out) {
  int idx = blockIdx.x * blockDim.x + threadIdx.x;
  float a = xl[idx], g = gate[idx];
  out[idx] = a + g * (xg[idx] - a);
}

// ---------------------------------------------------------------------------
extern "C" void kernel_launch(void* const* d_in, const int* in_sizes, int n_in,
                              void* d_out, int out_size, void* d_ws, size_t ws_size,
                              hipStream_t stream) {
  (void)in_sizes; (void)n_in; (void)out_size; (void)ws_size;
  const float* x   = (const float*)d_in[0];
  const float* lqw = (const float*)d_in[1];  const float* lqb = (const float*)d_in[2];
  const float* lkw = (const float*)d_in[3];  const float* lkb = (const float*)d_in[4];
  const float* lvw = (const float*)d_in[5];  const float* lvb = (const float*)d_in[6];
  const float* low = (const float*)d_in[7];  const float* lob = (const float*)d_in[8];
  const float* m1w = (const float*)d_in[9];  const float* m1b = (const float*)d_in[10];
  const float* m2w = (const float*)d_in[11]; const float* m2b = (const float*)d_in[12];
  const float* gqw = (const float*)d_in[13]; const float* gqb = (const float*)d_in[14];
  const float* gkw = (const float*)d_in[15]; const float* gkb = (const float*)d_in[16];
  const float* gvw = (const float*)d_in[17]; const float* gvb = (const float*)d_in[18];
  const float* gow = (const float*)d_in[19]; const float* gob = (const float*)d_in[20];
  const float* g1w = (const float*)d_in[21]; const float* g1b = (const float*)d_in[22];
  const float* g2w = (const float*)d_in[23]; const float* g2b = (const float*)d_in[24];

  float* W = (float*)d_ws;
  const size_t SZ = (size_t)BQ * NQ * DQ;   // 4M floats per big buffer
  const size_t SS = (size_t)BQ * CQ * DQ;   // 64K floats per small buffer
  float* hb0 = W;
  float* hb1 = W + SZ;
  float* qb  = W + 2 * SZ;
  float* kb  = W + 3 * SZ;
  float* vb  = W + 4 * SZ;
  float* aob = W + 5 * SZ;
  float* sm0 = W + 6 * SZ;
  float* reps = sm0;           float* mlin = sm0 + SS;
  float* mtmp = sm0 + 2 * SS;  float* rmix = sm0 + 3 * SS;
  float* gq   = sm0 + 4 * SS;  float* gk   = sm0 + 5 * SS;
  float* gv   = sm0 + 6 * SS;  float* gao  = sm0 + 7 * SS;
  float* rnew = sm0 + 8 * SS;

  dim3 blk(256);
  dim3 gB(DQ / 128, (BQ * NQ) / 128);  // 8 x 32 for big GEMMs
  dim3 gS(DQ / 128, 1);                // 8 x 1 for cluster GEMMs (M=64, clamped)

  // ---- T=2 local windowed-attention layers with residual ----
  const float* hin = x;
  float* houts[2] = {hb0, hb1};
  for (int t = 0; t < 2; ++t) {
    const float* wq = lqw + (size_t)t * DQ * DQ; const float* bq = lqb + (size_t)t * DQ;
    const float* wk = lkw + (size_t)t * DQ * DQ; const float* bk = lkb + (size_t)t * DQ;
    const float* wv = lvw + (size_t)t * DQ * DQ; const float* bv = lvb + (size_t)t * DQ;
    const float* wo = low + (size_t)t * DQ * DQ; const float* bo = lob + (size_t)t * DQ;
    gemm_kernel<<<gB, blk, 0, stream>>>(hin, nullptr, wq, bq, nullptr, qb,  BQ * NQ, DQ, DQ, 0);
    gemm_kernel<<<gB, blk, 0, stream>>>(hin, nullptr, wk, bk, nullptr, kb,  BQ * NQ, DQ, DQ, 0);
    gemm_kernel<<<gB, blk, 0, stream>>>(hin, nullptr, wv, bv, nullptr, vb,  BQ * NQ, DQ, DQ, 0);
    attn_kernel<<<dim3(NQ / 16, HQ, BQ), dim3(32), 0, stream>>>(qb, kb, vb, aob, NQ, 1);
    gemm_kernel<<<gB, blk, 0, stream>>>(aob, nullptr, wo, bo, hin, houts[t], BQ * NQ, DQ, DQ, 0);
    hin = houts[t];
  }
  const float* xl = hb1;

  // ---- cluster reps + cross-cluster MLP ----
  mean_kernel<<<(BQ * CQ * DQ) / 256, 256, 0, stream>>>(xl, reps);
  clustermix_kernel<<<(BQ * CQ * DQ) / 256, 256, 0, stream>>>(reps, mlin);
  gemm_kernel<<<gS, blk, 0, stream>>>(mlin, nullptr, m1w, m1b, nullptr, mtmp, BQ * CQ, DQ, DQ, 1);
  gemm_kernel<<<gS, blk, 0, stream>>>(mtmp, nullptr, m2w, m2b, reps,    rmix, BQ * CQ, DQ, DQ, 0);

  // ---- global cluster attention ----
  gemm_kernel<<<gS, blk, 0, stream>>>(rmix, nullptr, gqw, gqb, nullptr, gq, BQ * CQ, DQ, DQ, 0);
  gemm_kernel<<<gS, blk, 0, stream>>>(rmix, nullptr, gkw, gkb, nullptr, gk, BQ * CQ, DQ, DQ, 0);
  gemm_kernel<<<gS, blk, 0, stream>>>(rmix, nullptr, gvw, gvb, nullptr, gv, BQ * CQ, DQ, DQ, 0);
  attn_kernel<<<dim3(CQ / 16, HQ, BQ), dim3(32), 0, stream>>>(gq, gk, gv, gao, CQ, 0);
  gemm_kernel<<<gS, blk, 0, stream>>>(gao, nullptr, gow, gob, nullptr, rnew, BQ * CQ, DQ, DQ, 0);

  // ---- broadcast + gated merge ----
  bcast_kernel<<<(BQ * NQ * DQ) / 256, 256, 0, stream>>>(rnew, qb);   // qb := x_global
  // hidden = gelu(concat(x_local, x_global) @ g1w^T + g1b)  (split-K GEMM, Ktot=2048)
  gemm_kernel<<<gB, blk, 0, stream>>>(xl, qb, g1w, g1b, nullptr, kb, BQ * NQ, DQ, 2 * DQ, 1);
  // gate = sigmoid(hidden @ g2w^T + g2b)
  gemm_kernel<<<gB, blk, 0, stream>>>(kb, nullptr, g2w, g2b, nullptr, vb, BQ * NQ, DQ, DQ, 2);
  final_kernel<<<(BQ * NQ * DQ) / 256, 256, 0, stream>>>(xl, qb, vb, (float*)d_out);
}
